// MWDLSTMGCT_292057776706
// MI455X (gfx1250) — compile-verified, matrix operand order checked by reading, not run on hardware
//
#include <hip/hip_runtime.h>
#include <hip/hip_bf16.h>

// CDNA5 / gfx1250 wave32 WMMA types
typedef __attribute__((ext_vector_type(16))) __bf16 v16bf;
typedef __attribute__((ext_vector_type(8)))  float  v8f;

// Fast transcendentals: gfx1250 has single-instruction v_rcp_f32 / v_tanh_f32
// (TRANS ops co-execute with WMMA per ISA S_DELAY_ALU notes).
__device__ __forceinline__ float fast_rcp(float x) {
  return __builtin_amdgcn_rcpf(x);
}
__device__ __forceinline__ float sigf(float x) {
  return fast_rcp(1.0f + __expf(-x));       // v_exp_f32 + v_add + v_rcp
}
__device__ __forceinline__ float tanh_fast(float x) {
#if __has_builtin(__builtin_amdgcn_tanhf)
  return __builtin_amdgcn_tanhf(x);          // v_tanh_f32
#else
  return 1.0f - 2.0f * fast_rcp(__expf(2.0f * x) + 1.0f);
#endif
}

// Build a 16x32 bf16 WMMA A/B fragment from row-major fp32 memory.
// ISA 16-bit A layout (7.12.2): lanes 0-15 row=lane, K = k0+0..7 and k0+16..23;
// lanes 16-31 row=lane-16, K offset +8.  B(32x16) uses the same per-lane map
// with "row" = output column n, since B[k][n] = W[n][k] (row-major W, ld=K).
__device__ __forceinline__ v16bf frag_from_f32(const float* __restrict__ base,
                                               int ld, int rc0, int k0) {
  const int lane = threadIdx.x & 31;
  const float* q = base + (size_t)(rc0 + (lane & 15)) * ld
                        + (k0 + ((lane >> 4) << 3));
  v16bf r;
#pragma unroll
  for (int i = 0; i < 8; ++i) { r[i] = (__bf16)q[i]; r[8 + i] = (__bf16)q[16 + i]; }
  return r;
}

// out(M x N/2) = avgpool2_N( sigmoid( A(MxK) @ W(NxK)^T + bias(N) ) )
// One wave computes a 16(M) x 64(N) register tile.  All four B fragments are
// converted into distinct registers BEFORE the four back-to-back WMMAs, so the
// bf16 WMMA->VALU WAR hazard slots are filled with WMMAs instead of v_nops.
__global__ __launch_bounds__(32) void gemm_sig_pool(const float* __restrict__ A,
                                                    const float* __restrict__ W,
                                                    const float* __restrict__ bias,
                                                    float* __restrict__ out,
                                                    int K, int N) {
  const int mt  = blockIdx.x;
  const int nt0 = blockIdx.y * 4;
  const int lane = threadIdx.x & 31;
  v8f acc[4] = {};
  for (int k0 = 0; k0 < K; k0 += 32) {
    v16bf a  = frag_from_f32(A, K, mt * 16, k0);
    v16bf b0 = frag_from_f32(W, K, (nt0 + 0) * 16, k0);
    v16bf b1 = frag_from_f32(W, K, (nt0 + 1) * 16, k0);
    v16bf b2 = frag_from_f32(W, K, (nt0 + 2) * 16, k0);
    v16bf b3 = frag_from_f32(W, K, (nt0 + 3) * 16, k0);
    acc[0] = __builtin_amdgcn_wmma_f32_16x16x32_bf16(false, a, false, b0,
                                                     (short)0, acc[0], false, false);
    acc[1] = __builtin_amdgcn_wmma_f32_16x16x32_bf16(false, a, false, b1,
                                                     (short)0, acc[1], false, false);
    acc[2] = __builtin_amdgcn_wmma_f32_16x16x32_bf16(false, a, false, b2,
                                                     (short)0, acc[2], false, false);
    acc[3] = __builtin_amdgcn_wmma_f32_16x16x32_bf16(false, a, false, b3,
                                                     (short)0, acc[3], false, false);
  }
  // C layout: VGPR v -> M = mt*16 + v (+8 for lanes 16-31); N = nt*16 + (lane&15)
  const int col  = lane & 15;
  const int half = N >> 1;
#pragma unroll
  for (int j = 0; j < 4; ++j) {
    const int nt = nt0 + j;
    const float bn = bias[nt * 16 + col];
#pragma unroll
    for (int v = 0; v < 8; ++v) {
      float s = sigf(acc[j][v] + bn);
      float p = 0.5f * (s + __shfl_xor(s, 1, 32));   // avgpool2 over adjacent N
      if ((lane & 1) == 0) {
        int row = mt * 16 + v + ((lane >> 4) << 3);
        out[(size_t)row * half + nt * 8 + (col >> 1)] = p;
      }
    }
  }
}

// Pre-swizzle Whh3 (400x100 fp32, row-major) into bf16 WMMA B fragments,
// K zero-padded to 128: 25 n-tiles x 4 k-steps x 32 lanes x 16 halves.
__global__ void prep_whh_frags(const float* __restrict__ Whh,
                               unsigned short* __restrict__ frag_u16) {
  __bf16* frag = reinterpret_cast<__bf16*>(frag_u16);
  int idx = blockIdx.x * blockDim.x + threadIdx.x;     // [0, 25*4*32)
  if (idx >= 25 * 4 * 32) return;
  int lane = idx & 31, s = (idx >> 5) & 3, nt = idx >> 7;
  int n  = nt * 16 + (lane & 15);
  int kb = s * 32 + ((lane >> 4) << 3);
  __bf16* d = frag + (size_t)idx * 16;
#pragma unroll
  for (int i = 0; i < 8; ++i) {
    int k1 = kb + i, k2 = kb + 16 + i;
    d[i]     = (k1 < 100) ? (__bf16)Whh[n * 100 + k1] : (__bf16)0.0f;
    d[8 + i] = (k2 < 100) ? (__bf16)Whh[n * 100 + k2] : (__bf16)0.0f;
  }
}

// Persistent LSTM over 256 timesteps. One block = 16 batch rows, 8 waves.
// State (h fp32 + bf16 copy, c) and the 16x400 gate buffer live in LDS.
// Per step: z = gx + h @ Whh^T via 25 gate-tiles x 4 bf16 WMMAs, then cell update.
__global__ __launch_bounds__(256) void lstm_tail(
    const float* __restrict__ xl2,            // 256 x 256
    const unsigned short* __restrict__ whhfrag_u16,
    const float* __restrict__ Wih,            // 400 (Wih3[:,0])
    const float* __restrict__ bg,             // 400
    const float* __restrict__ Wout,           // 100
    const float* __restrict__ bout,           // 1
    float* __restrict__ out)                  // 256
{
  const __bf16* whhfrag = reinterpret_cast<const __bf16*>(whhfrag_u16);
  __shared__ float  zly[16][400];
  __shared__ float  cst[16][100];
  __shared__ float  hf [16][100];
  __shared__ __bf16 hbf[16][128];             // zero-padded K for WMMA
  __shared__ float  xts[16];
  __shared__ float  wihs[400], bgs[400], wouts[100];

  const int tid  = threadIdx.x;
  const int lane = tid & 31;
  const int wave = tid >> 5;
  const int btile = blockIdx.x;

  for (int i = tid; i < 400; i += 256) { wihs[i] = Wih[i]; bgs[i] = bg[i]; }
  for (int i = tid; i < 100; i += 256) wouts[i] = Wout[i];
  for (int i = tid; i < 16 * 128; i += 256) (&hbf[0][0])[i] = (__bf16)0.0f;
  for (int i = tid; i < 16 * 100; i += 256) { (&cst[0][0])[i] = 0.0f; (&hf[0][0])[i] = 0.0f; }
  __syncthreads();

  for (int t = 0; t < 256; ++t) {
    if (tid < 16) xts[tid] = xl2[(size_t)(btile * 16 + tid) * 256 + t];
    __syncthreads();

    // A fragments (shared by all 25 gate tiles): h from LDS, K-steps 0..3
    v16bf afr[4];
#pragma unroll
    for (int s = 0; s < 4; ++s) {
      const __bf16* q = &hbf[lane & 15][s * 32 + ((lane >> 4) << 3)];
#pragma unroll
      for (int i = 0; i < 8; ++i) { afr[s][i] = q[i]; afr[s][8 + i] = q[16 + i]; }
    }

    for (int ntile = wave; ntile < 25; ntile += 8) {
      // Load all 4 K-step B fragments first, then chain 4 WMMAs back-to-back
      // (accumulation on C/D is the designed dependence; no hazard NOPs).
      v16bf bfr[4];
#pragma unroll
      for (int s = 0; s < 4; ++s)
        bfr[s] = *(const v16bf*)(whhfrag +
                   (size_t)(((ntile * 4 + s) * 32) + lane) * 16);
      v8f acc = {};
#pragma unroll
      for (int s = 0; s < 4; ++s)
        acc = __builtin_amdgcn_wmma_f32_16x16x32_bf16(false, afr[s], false, bfr[s],
                                                      (short)0, acc, false, false);
      const int col = lane & 15;
      const int g = ntile * 16 + col;
      const float wg = wihs[g], bb = bgs[g];
#pragma unroll
      for (int v = 0; v < 8; ++v) {
        int r = v + ((lane >> 4) << 3);
        zly[r][g] = acc[v] + xts[r] * wg + bb;   // z = h@Whh^T + x*Wih + b
      }
    }
    __syncthreads();

    // cell update: 16 rows x 100 hidden (exp/rcp/tanh single TRANS ops)
    for (int i = tid; i < 1600; i += 256) {
      int r = i / 100, hh = i - r * 100;
      float zi = zly[r][hh],       zf = zly[r][100 + hh];
      float zg = zly[r][200 + hh], zo = zly[r][300 + hh];
      float c = sigf(zf) * cst[r][hh] + sigf(zi) * tanh_fast(zg);
      cst[r][hh] = c;
      float h = sigf(zo) * tanh_fast(c);
      hf[r][hh] = h;
      hbf[r][hh] = (__bf16)h;
    }
    __syncthreads();
  }

  // out[b] = h_last . Wout + bout   (only s3's last step survives the concat)
  if (tid < 16) {
    float s = bout[0];
    for (int k = 0; k < 100; ++k) s += hf[tid][k] * wouts[k];
    out[btile * 16 + tid] = s;
  }
}

extern "C" void kernel_launch(void* const* d_in, const int* in_sizes, int n_in,
                              void* d_out, int out_size, void* d_ws, size_t ws_size,
                              hipStream_t stream) {
  (void)in_sizes; (void)n_in; (void)out_size; (void)ws_size;
  const float* x    = (const float*)d_in[0];   // (256,1024,1) -> inp 256x1024
  const float* W1L  = (const float*)d_in[3];
  const float* b1L  = (const float*)d_in[4];
  const float* W2L  = (const float*)d_in[7];
  const float* b2L  = (const float*)d_in[8];
  const float* Wih3 = (const float*)d_in[15];
  const float* Whh3 = (const float*)d_in[16];
  const float* b3   = (const float*)d_in[17];
  const float* Wout = (const float*)d_in[18];
  const float* bout = (const float*)d_in[19];
  float* out = (float*)d_out;

  char* ws = (char*)d_ws;
  float*          xl1     = (float*)ws;                               // 256*512 f32
  float*          xl2     = (float*)(ws + 256 * 512 * 4);             // 256*256 f32
  unsigned short* whhfrag = (unsigned short*)(ws + 256 * 512 * 4
                                                 + 256 * 256 * 4);    // 51200 bf16

  // One-time (per launch) bf16 fragment swizzle of Whh3
  prep_whh_frags<<<dim3(13), dim3(256), 0, stream>>>(Whh3, whhfrag);
  // Layer 1: 256x1024 @ 1024x1024^T, fused sigmoid+pool -> xl1 (256x512)
  gemm_sig_pool<<<dim3(16, 16), dim3(32), 0, stream>>>(x, W1L, b1L, xl1, 1024, 1024);
  // Layer 2: 256x512 @ 512x512^T, fused sigmoid+pool -> xl2 (256x256)
  gemm_sig_pool<<<dim3(16, 8), dim3(32), 0, stream>>>(xl1, W2L, b2L, xl2, 512, 512);
  // Persistent LSTM (only the surviving branch) + output projection
  lstm_tail<<<dim3(16), dim3(256), 0, stream>>>(xl2, whhfrag, Wih3, b3, Wout, bout, out);
}
